// Net_68753836474417
// MI455X (gfx1250) — matrix operand order repaired and checked by reading, not compiled, not run on hardware
//
#include <hip/hip_runtime.h>
#include <hip/hip_bf16.h>
#include <stdint.h>

typedef __attribute__((ext_vector_type(16))) _Float16 v16h;
typedef __attribute__((ext_vector_type(8)))  float    v8f;

union FragU { v16h v; uint4 q[2]; };
union PairU { uint32_t u; _Float16 h[2]; };

// ---------------------------------------------------------------------------
// Pack W (row-major [N][K], f32) into per-lane WMMA B-fragment order, f16.
// Fragment dword r of lane L for tile (nt,kt):
//   n = nt*16 + (L&15), hi = L>>4, k = kt*32 + (r>>2)*16 + hi*8 + (r&3)*2  (pair k, k+1)
// Out index: ((nt*num_kt + kt)*32 + lane)*8 + r   (one v16h per lane per tile)
// Out-of-range (N,K) entries are zero => padding is exact through relu chain.
// ---------------------------------------------------------------------------
__global__ void pack_weights_f16(const float* __restrict__ W, uint32_t* __restrict__ out,
                                 int N, int K, int num_nt, int num_kt) {
    int idx = blockIdx.x * blockDim.x + threadIdx.x;
    int total = num_nt * num_kt * 32 * 8;
    if (idx >= total) return;
    int r    = idx & 7;
    int lane = (idx >> 3) & 31;
    int t    = idx >> 8;
    int kt   = t % num_kt;
    int nt   = t / num_kt;
    int n  = nt * 16 + (lane & 15);
    int hi = (lane >> 4) & 1;
    int k  = kt * 32 + (r >> 2) * 16 + hi * 8 + (r & 3) * 2;
    float v0 = (n < N && k     < K) ? W[n * K + k]     : 0.f;
    float v1 = (n < N && k + 1 < K) ? W[n * K + k + 1] : 0.f;
    PairU p; p.h[0] = (_Float16)v0; p.h[1] = (_Float16)v1;
    out[idx] = p.u;
}

// ---------------------------------------------------------------------------
// Conv1 as implicit GEMM: [576 pixels, 25->32] x [32, 6->16] per image.
// One block (128 thr = 4 waves) per image. Each wave does 9 M-tiles, 1 WMMA
// each. A-fragments gathered per-lane from LDS image via ktab offsets
// (ktab[k] = kh*28 + kw, invalid k -> zero pad zone). Then pool+bias+relu.
// Output: f16 act1[img][6][12][12].
// ---------------------------------------------------------------------------
__global__ void conv1_wmma(const float* __restrict__ x, const uint32_t* __restrict__ w1c,
                           const float* __restrict__ cb1, _Float16* __restrict__ act1) {
    __shared__ _Float16 xs[1472];            // 784 image + zero pad zone
    __shared__ unsigned short ktab[32];
    __shared__ _Float16 conv[576 * 6];       // conv outputs [m][c], pre-bias
    long img = blockIdx.x;
    int tid = threadIdx.x;
    const int wave = tid >> 5;
    const int lane = tid & 31;
    const int col  = lane & 15;
    const int hi   = lane >> 4;

    for (int i = tid; i < 784; i += 128) xs[i] = (_Float16)x[img * 784 + i];
    for (int i = 784 + tid; i < 1472; i += 128) xs[i] = (_Float16)0.f;
    if (tid < 32) ktab[tid] = (tid < 25) ? (unsigned short)((tid / 5) * 28 + tid % 5)
                                         : (unsigned short)800;  // zero zone
    __syncthreads();

    v16h b1 = ((const v16h*)w1c)[lane];      // single B fragment (nt=0, kt=0)
    for (int t = wave; t < 36; t += 4) {     // uniform per wave: 9 tiles each
        int m  = t * 16 + col;               // this lane's A row (pixel)
        int oh = m / 24, ow = m % 24;
        int base = oh * 28 + ow;
        v16h a;
#pragma unroll
        for (int r = 0; r < 8; ++r) {
            int k0 = ((r >> 2) * 16) + hi * 8 + (r & 3) * 2;
            a[2 * r]     = xs[ktab[k0]     + base];
            a[2 * r + 1] = xs[ktab[k0 + 1] + base];
        }
        v8f c = {};
        c = __builtin_amdgcn_wmma_f32_16x16x32_f16(false, a, false, b1,
                                                   (short)0, c, false, false);
        if (col < 6) {
#pragma unroll
            for (int r = 0; r < 8; ++r)
                conv[(t * 16 + r + 8 * hi) * 6 + col] = (_Float16)c[r];
        }
    }
    __syncthreads();

    // 2x2 maxpool + bias + relu; pooled index o = c*144 + ph*12 + pw
    for (int o = tid; o < 864; o += 128) {
        int c  = o / 144;
        int p  = o % 144;
        int ph = p / 12, pw = p % 12;
        float mx = -3.4e38f;
#pragma unroll
        for (int dy = 0; dy < 2; ++dy)
#pragma unroll
            for (int dx = 0; dx < 2; ++dx)
                mx = fmaxf(mx, (float)conv[((2 * ph + dy) * 24 + (2 * pw + dx)) * 6 + c]);
        act1[img * 864 + o] = (_Float16)fmaxf(mx + cb1[c], 0.f);
    }
}

// ---------------------------------------------------------------------------
// Conv2 as implicit GEMM: [64 pixels, 150->160] x [160, 16] per image.
// One block (128 thr = 4 waves) per image; wave w owns M-tile w (16 pixels),
// 5 K-chunk WMMAs. ktab[k] = ic*144 + kh*12 + kw (invalid -> zero zone).
// Then pool+bias+relu -> f16 act2[img][256] in reshape order oc*16+ph*4+pw.
// ---------------------------------------------------------------------------
__global__ void conv2_wmma(const _Float16* __restrict__ act1, const uint32_t* __restrict__ w2c,
                           const float* __restrict__ cb2, _Float16* __restrict__ act2) {
    __shared__ _Float16 a1s[1024];           // 864 acts + zero pad zone
    __shared__ unsigned short ktab[160];
    __shared__ float convb[64 * 16];         // conv outputs [m][oc], pre-bias
    long img = blockIdx.x;
    int tid = threadIdx.x;
    const int wave = tid >> 5;
    const int lane = tid & 31;
    const int col  = lane & 15;
    const int hi   = lane >> 4;

    for (int i = tid; i < 864; i += 128) a1s[i] = act1[img * 864 + i];
    for (int i = 864 + tid; i < 1024; i += 128) a1s[i] = (_Float16)0.f;
    if (tid < 160) {
        int k = tid;
        if (k < 150) {
            int ic = k / 25, rem = k % 25;
            ktab[k] = (unsigned short)(ic * 144 + (rem / 5) * 12 + rem % 5);
        } else {
            ktab[k] = (unsigned short)864;   // zero zone
        }
    }
    __syncthreads();

    int m  = wave * 16 + col;                // this lane's A row (pixel)
    int oh = m >> 3, ow = m & 7;
    int base = oh * 12 + ow;
    const v16h* w2v = (const v16h*)w2c;
    v8f c = {};
#pragma unroll
    for (int kc = 0; kc < 5; ++kc) {
        v16h a;
#pragma unroll
        for (int r = 0; r < 8; ++r) {
            int k0 = kc * 32 + ((r >> 2) * 16) + hi * 8 + (r & 3) * 2;
            a[2 * r]     = a1s[ktab[k0]     + base];
            a[2 * r + 1] = a1s[ktab[k0 + 1] + base];
        }
        v16h b = w2v[kc * 32 + lane];
        c = __builtin_amdgcn_wmma_f32_16x16x32_f16(false, a, false, b,
                                                   (short)0, c, false, false);
    }
#pragma unroll
    for (int r = 0; r < 8; ++r)
        convb[(wave * 16 + r + 8 * hi) * 16 + col] = c[r];
    __syncthreads();

    // 2x2 maxpool + bias + relu; output index o = oc*16 + ph*4 + pw
    for (int o = tid; o < 256; o += 128) {
        int oc = o >> 4;
        int ph = (o >> 2) & 3, pw = o & 3;
        float mx = -3.4e38f;
#pragma unroll
        for (int dy = 0; dy < 2; ++dy)
#pragma unroll
            for (int dx = 0; dx < 2; ++dx)
                mx = fmaxf(mx, convb[((2 * ph + dy) * 8 + (2 * pw + dx)) * 16 + oc]);
        act2[img * 256 + o] = (_Float16)fmaxf(mx + cb2[oc], 0.f);
    }
}

// ---------------------------------------------------------------------------
// FC1(256->120) + FC2(120->84) + FC3(84->10) + softmax, all WMMA f16->f32.
// 256 threads = 8 waves; each wave owns a 16-image tile and a private 4KB LDS
// slice. A-fragments: ISA 16-bit A layout (K pairs contiguous -> b128 loads).
// B-fragments: one contiguous v16h per lane from the packed weight buffers.
// ---------------------------------------------------------------------------
__global__ void fc_softmax_wmma(const _Float16* __restrict__ act2,
                                const uint32_t* __restrict__ w1p, const float* __restrict__ fb1,
                                const uint32_t* __restrict__ w2p, const float* __restrict__ fb2,
                                const uint32_t* __restrict__ w3p, const float* __restrict__ fb3,
                                float* __restrict__ probs, int nimg) {
    __shared__ _Float16 lds[8][16 * 128];   // 32 KB, one [16][128] slice per wave
    const int wave = threadIdx.x >> 5;
    const int lane = threadIdx.x & 31;
    const int col  = lane & 15;             // A: row m ; B/C/D: column n
    const int hi   = lane >> 4;
    _Float16* actw = &lds[wave][0];
    long img_base = ((long)blockIdx.x * 8 + wave) * 16;

    __builtin_prefetch(w1p, 0, 3);          // global_prefetch_b8: warm weight tiles

    // ---- FC1: [16,256] x [256,128]  (8 K-chunks, 8 N-tiles) ----
    v16h A1[8];
    {
        long imgr = img_base + col; if (imgr > nimg - 1) imgr = nimg - 1;
        const uint4* arow = (const uint4*)(act2 + imgr * 256);
#pragma unroll
        for (int kc = 0; kc < 8; ++kc) {
            FragU f; f.q[0] = arow[kc * 4 + hi]; f.q[1] = arow[kc * 4 + hi + 2];
            A1[kc] = f.v;
        }
    }
    const v16h* w1v = (const v16h*)w1p;
    for (int nt = 0; nt < 8; ++nt) {
        v8f c = {};
#pragma unroll
        for (int kt = 0; kt < 8; ++kt) {
            v16h b = w1v[(nt * 8 + kt) * 32 + lane];
            c = __builtin_amdgcn_wmma_f32_16x16x32_f16(false, A1[kt], false, b,
                                                       (short)0, c, false, false);
        }
        int n = nt * 16 + col;
        float bias = (n < 120) ? fb1[n] : 0.f;
#pragma unroll
        for (int r = 0; r < 8; ++r)
            actw[(r + 8 * hi) * 128 + n] = (_Float16)fmaxf(c[r] + bias, 0.f);
    }
    __syncthreads();

    // ---- FC2: [16,128] x [128,96]  (4 K-chunks, 6 N-tiles) ----
    v16h A2[4];
    {
        const uint4* lrow = (const uint4*)(actw + col * 128);
#pragma unroll
        for (int kc = 0; kc < 4; ++kc) {
            FragU f; f.q[0] = lrow[kc * 4 + hi]; f.q[1] = lrow[kc * 4 + hi + 2];
            A2[kc] = f.v;
        }
    }
    __syncthreads();
    const v16h* w2v = (const v16h*)w2p;
    for (int nt = 0; nt < 6; ++nt) {
        v8f c = {};
#pragma unroll
        for (int kt = 0; kt < 4; ++kt) {
            v16h b = w2v[(nt * 4 + kt) * 32 + lane];
            c = __builtin_amdgcn_wmma_f32_16x16x32_f16(false, A2[kt], false, b,
                                                       (short)0, c, false, false);
        }
        int n = nt * 16 + col;
        float bias = (n < 84) ? fb2[n] : 0.f;
#pragma unroll
        for (int r = 0; r < 8; ++r)
            actw[(r + 8 * hi) * 128 + n] = (_Float16)fmaxf(c[r] + bias, 0.f);
    }
    __syncthreads();

    // ---- FC3: [16,96] x [96,16]  (3 K-chunks, 1 N-tile) ----
    v16h A3[3];
    {
        const uint4* lrow = (const uint4*)(actw + col * 128);
#pragma unroll
        for (int kc = 0; kc < 3; ++kc) {
            FragU f; f.q[0] = lrow[kc * 4 + hi]; f.q[1] = lrow[kc * 4 + hi + 2];
            A3[kc] = f.v;
        }
    }
    __syncthreads();
    const v16h* w3v = (const v16h*)w3p;
    v8f c3 = {};
#pragma unroll
    for (int kt = 0; kt < 3; ++kt) {
        v16h b = w3v[kt * 32 + lane];
        c3 = __builtin_amdgcn_wmma_f32_16x16x32_f16(false, A3[kt], false, b,
                                                    (short)0, c3, false, false);
    }
    float bias3 = (col < 10) ? fb3[col] : 0.f;
    float* lg = (float*)actw;               // reuse slice as [16][16] f32 logits
#pragma unroll
    for (int r = 0; r < 8; ++r)
        lg[(r + 8 * hi) * 16 + col] = c3[r] + bias3;
    __syncthreads();

    // ---- per-row softmax (10 classes), lanes 0..15 each own one image ----
    if (lane < 16) {
        long img = img_base + lane;
        if (img < nimg) {
            float v[10], mx = -3.4e38f;
#pragma unroll
            for (int j = 0; j < 10; ++j) { v[j] = lg[lane * 16 + j]; mx = fmaxf(mx, v[j]); }
            float s = 0.f;
#pragma unroll
            for (int j = 0; j < 10; ++j) { v[j] = __expf(v[j] - mx); s += v[j]; }
            float inv = 1.f / s;
#pragma unroll
            for (int j = 0; j < 10; ++j) probs[img * 10 + j] = v[j] * inv;
        }
    }
}

// ---------------------------------------------------------------------------
// z[t] = sum_j p1[j]*p2[t-j]  (full convolution, length 19), out = log(z)
// ---------------------------------------------------------------------------
__global__ void pair_conv_log(const float* __restrict__ probs, float* __restrict__ out, int B) {
    long b = (long)blockIdx.x * blockDim.x + threadIdx.x;
    if (b >= B) return;
    const float* p1 = probs + (2 * b) * 10;
    const float* p2 = p1 + 10;
    float a[10], q[10];
#pragma unroll
    for (int j = 0; j < 10; ++j) { a[j] = p1[j]; q[j] = p2[j]; }
#pragma unroll
    for (int t = 0; t < 19; ++t) {
        float z = 0.f;
        int jlo = t - 9 > 0 ? t - 9 : 0;
        int jhi = t < 9 ? t : 9;
        for (int j = jlo; j <= jhi; ++j) z = fmaf(a[j], q[t - j], z);
        out[b * 19 + t] = __logf(z);
    }
}

// ---------------------------------------------------------------------------
extern "C" void kernel_launch(void* const* d_in, const int* in_sizes, int n_in,
                              void* d_out, int out_size, void* d_ws, size_t ws_size,
                              hipStream_t stream) {
    const float* x   = (const float*)d_in[0];
    const float* cw1 = (const float*)d_in[1];
    const float* cb1 = (const float*)d_in[2];
    const float* cw2 = (const float*)d_in[3];
    const float* cb2 = (const float*)d_in[4];
    const float* fw1 = (const float*)d_in[5];
    const float* fb1 = (const float*)d_in[6];
    const float* fw2 = (const float*)d_in[7];
    const float* fb2 = (const float*)d_in[8];
    const float* fw3 = (const float*)d_in[9];
    const float* fb3 = (const float*)d_in[10];

    const long B    = (long)in_sizes[0] / (2 * 28 * 28);
    const long NIMG = 2 * B;

    char* ws = (char*)d_ws;
    size_t off = 0;
    auto alloc = [&](size_t bytes) {
        void* p = ws + off;
        off = (off + bytes + 255) & ~(size_t)255;
        return p;
    };
    _Float16* act1  = (_Float16*)alloc((size_t)NIMG * 864 * sizeof(_Float16)); // ~56.6 MB
    _Float16* act2  = (_Float16*)alloc((size_t)NIMG * 256 * sizeof(_Float16)); // ~16.8 MB
    float*    probs = (float*)   alloc((size_t)NIMG * 10 * sizeof(float));     // ~1.3 MB
    uint32_t* w1p   = (uint32_t*)alloc(8 * 8 * 256 * 4);  // fc1 B-frags
    uint32_t* w2p   = (uint32_t*)alloc(6 * 4 * 256 * 4);  // fc2 B-frags
    uint32_t* w3p   = (uint32_t*)alloc(1 * 3 * 256 * 4);  // fc3 B-frags
    uint32_t* w1c   = (uint32_t*)alloc(1 * 1 * 256 * 4);  // conv1 B-frags [6][25]
    uint32_t* w2c   = (uint32_t*)alloc(1 * 5 * 256 * 4);  // conv2 B-frags [16][150]

    pack_weights_f16<<<(8 * 8 * 256 + 255) / 256, 256, 0, stream>>>(fw1, w1p, 120, 256, 8, 8);
    pack_weights_f16<<<(6 * 4 * 256 + 255) / 256, 256, 0, stream>>>(fw2, w2p,  84, 120, 6, 4);
    pack_weights_f16<<<(1 * 3 * 256 + 255) / 256, 256, 0, stream>>>(fw3, w3p,  10,  84, 1, 3);
    pack_weights_f16<<<1, 256, 0, stream>>>(cw1, w1c,  6,  25, 1, 1);
    pack_weights_f16<<<(1 * 5 * 256 + 255) / 256, 256, 0, stream>>>(cw2, w2c, 16, 150, 1, 5);

    conv1_wmma<<<NIMG, 128, 0, stream>>>(x, w1c, cb1, act1);
    conv2_wmma<<<NIMG, 128, 0, stream>>>(act1, w2c, cb2, act2);

    long tiles  = (NIMG + 15) / 16;
    long blocks = (tiles + 7) / 8;
    fc_softmax_wmma<<<blocks, 256, 0, stream>>>(act2, w1p, fb1, w2p, fb2, w3p, fb3,
                                                probs, (int)NIMG);
    pair_conv_log<<<(B + 255) / 256, 256, 0, stream>>>(probs, (float*)d_out, (int)B);
}
